// Lane2Lane_7653631721783
// MI455X (gfx1250) — matrix operand order repaired and checked by reading, not compiled, not run on hardware
//
#include <hip/hip_runtime.h>

#define N_NODES 50000
#define DDIM    256
#define SHOPS   6
#define RROUNDS 4
#define MTILES  (N_NODES / 16)   // 3125 (exact)

typedef __bf16 v8bf  __attribute__((ext_vector_type(8)));
typedef __bf16 v16bf __attribute__((ext_vector_type(16)));
typedef float  v8f   __attribute__((ext_vector_type(8)));

// fp32 -> bf16 (round to nearest even), raw ushort storage
__device__ __forceinline__ unsigned short f2bf_raw(float f) {
    union { float f; unsigned u; } c; c.f = f;
    unsigned r = c.u + 0x7FFFu + ((c.u >> 16) & 1u);
    return (unsigned short)(r >> 16);
}

// ---------------------------------------------------------------------------
// GEMM: out[N,256] = A[N,256] @ W[256,256]^T  in bf16 with f32 accumulation.
//   gidx != nullptr : A row m is xb[gidx[m]] (gather)
//   sidx != nullptr : out row m is atomically accumulated into out[sidx[m]]
// Block = 256 threads = 8 waves. Wave w: M-tile = blockIdx.x*4 + (w&3),
// N-half = (w>>2)*128.  Each wave: 16x128 output, K-loop 8 x wmma over 8 N-tiles.
// ---------------------------------------------------------------------------
__global__ __launch_bounds__(256)
void gemm_bf16(const unsigned short* __restrict__ xb,
               const unsigned short* __restrict__ w,
               float* __restrict__ out,
               const int* __restrict__ gidx,
               const int* __restrict__ sidx)
{
    const int lane  = threadIdx.x & 31;
    const int wave  = threadIdx.x >> 5;
    const int mtile = blockIdx.x * 4 + (wave & 3);
    if (mtile >= MTILES) return;                 // wave-uniform: EXEC stays all-1s
    const int nbase = (wave >> 2) * 128;
    const int hi    = lane >> 4;                 // lane-half select
    const int l15   = lane & 15;

    int arow = mtile * 16 + l15;                 // A: M = lane%16
    if (gidx) arow = gidx[arow];
    const unsigned short* aptr = xb + (size_t)arow * DDIM;

    v8f acc[8] = {};

    for (int k0 = 0; k0 < DDIM; k0 += 32) {
        // A frag (16x32 bf16): lanes0-15 K={0..15}, lanes16-31 K={16..31}
        const int ka = k0 + hi * 16;
        v8bf alo = *(const v8bf*)(aptr + ka);
        v8bf ahi = *(const v8bf*)(aptr + ka + 8);
        v16bf af = __builtin_shufflevector(alo, ahi,
                    0,1,2,3,4,5,6,7,8,9,10,11,12,13,14,15);
        #pragma unroll
        for (int t = 0; t < 8; ++t) {
            // B frag (32x16 bf16): lane = column n, contiguous K per lane-half.
            // B[k,n] = W[n,k]  ->  contiguous 32B slice of row n of W. No transpose.
            const unsigned short* bptr = w + (size_t)(nbase + t * 16 + l15) * DDIM;
            v16bf bfr = *(const v16bf*)(bptr + k0 + hi * 16);
            acc[t] = __builtin_amdgcn_wmma_f32_16x16x32_bf16(
                         false, af, false, bfr, (short)0, acc[t], false, false);
        }
    }

    // C/D layout: lanes0-15 N=lane, M=r; lanes16-31 N=lane-16, M=r+8
    const int mrow = mtile * 16 + hi * 8;
    if (sidx) {
        int orow[8];
        #pragma unroll
        for (int r = 0; r < 8; ++r) orow[r] = sidx[mrow + r];
        #pragma unroll
        for (int t = 0; t < 8; ++t) {
            const int col = nbase + t * 16 + l15;
            #pragma unroll
            for (int r = 0; r < 8; ++r)
                __hip_atomic_fetch_add(&out[(size_t)orow[r] * DDIM + col],
                                       acc[t][r], __ATOMIC_RELAXED,
                                       __HIP_MEMORY_SCOPE_AGENT);
        }
    } else {
        #pragma unroll
        for (int t = 0; t < 8; ++t) {
            const int col = nbase + t * 16 + l15;
            #pragma unroll
            for (int r = 0; r < 8; ++r)
                out[(size_t)(mrow + r) * DDIM + col] = acc[t][r];
        }
    }
}

// ---------------------------------------------------------------------------
// GroupNorm(1 group) + ReLU over one row per wave; writes bf16 activations.
// ---------------------------------------------------------------------------
__global__ __launch_bounds__(256)
void gn_relu_bf16(const float* __restrict__ t, const float* __restrict__ w,
                  const float* __restrict__ b, unsigned short* __restrict__ xb)
{
    const int row  = blockIdx.x * 8 + (threadIdx.x >> 5);
    const int lane = threadIdx.x & 31;
    if (row >= N_NODES) return;
    const float* tr = t + (size_t)row * DDIM;
    float v[8]; float s = 0.f, ss = 0.f;
    #pragma unroll
    for (int j = 0; j < 8; ++j) {
        float x = tr[lane + 32 * j]; v[j] = x; s += x; ss += x * x;
    }
    #pragma unroll
    for (int off = 16; off > 0; off >>= 1) {
        s  += __shfl_xor(s,  off, 32);
        ss += __shfl_xor(ss, off, 32);
    }
    const float mean = s * (1.f / DDIM);
    const float rstd = rsqrtf(ss * (1.f / DDIM) - mean * mean + 1e-5f);
    #pragma unroll
    for (int j = 0; j < 8; ++j) {
        const int col = lane + 32 * j;
        float y = (v[j] - mean) * rstd * w[col] + b[col];
        y = fmaxf(y, 0.f);
        xb[(size_t)row * DDIM + col] = f2bf_raw(y);
    }
}

// ---------------------------------------------------------------------------
// GroupNorm + residual + ReLU; updates x (fp32, in d_out) and its bf16 mirror.
// ---------------------------------------------------------------------------
__global__ __launch_bounds__(256)
void gn_res_relu(const float* __restrict__ t, const float* __restrict__ w,
                 const float* __restrict__ b, float* __restrict__ x,
                 unsigned short* __restrict__ xb)
{
    const int row  = blockIdx.x * 8 + (threadIdx.x >> 5);
    const int lane = threadIdx.x & 31;
    if (row >= N_NODES) return;
    const float* tr = t + (size_t)row * DDIM;
    float v[8]; float s = 0.f, ss = 0.f;
    #pragma unroll
    for (int j = 0; j < 8; ++j) {
        float q = tr[lane + 32 * j]; v[j] = q; s += q; ss += q * q;
    }
    #pragma unroll
    for (int off = 16; off > 0; off >>= 1) {
        s  += __shfl_xor(s,  off, 32);
        ss += __shfl_xor(ss, off, 32);
    }
    const float mean = s * (1.f / DDIM);
    const float rstd = rsqrtf(ss * (1.f / DDIM) - mean * mean + 1e-5f);
    #pragma unroll
    for (int j = 0; j < 8; ++j) {
        const int col = lane + 32 * j;
        const size_t idx = (size_t)row * DDIM + col;
        float y = (v[j] - mean) * rstd * w[col] + b[col] + x[idx];
        y = fmaxf(y, 0.f);
        x[idx]  = y;
        xb[idx] = f2bf_raw(y);
    }
}

// ---------------------------------------------------------------------------
// One-time converts
// ---------------------------------------------------------------------------
__global__ void cvt_bf16(const float* __restrict__ in,
                         unsigned short* __restrict__ out, int n)
{
    int i = blockIdx.x * 256 + threadIdx.x;
    if (i < n) out[i] = f2bf_raw(in[i]);
}

__global__ void init_x(const float* __restrict__ in, float* __restrict__ x,
                       unsigned short* __restrict__ xb, int n)
{
    int i = blockIdx.x * 256 + threadIdx.x;
    if (i < n) { float v = in[i]; x[i] = v; xb[i] = f2bf_raw(v); }
}

// ---------------------------------------------------------------------------
extern "C" void kernel_launch(void* const* d_in, const int* in_sizes, int n_in,
                              void* d_out, int out_size, void* d_ws, size_t ws_size,
                              hipStream_t stream)
{
    const float* lane_features = (const float*)d_in[0];
    const float* W_ctr   = (const float*)d_in[1];
    const float* norm_w  = (const float*)d_in[2];
    const float* norm_b  = (const float*)d_in[3];
    const float* W_ctr2  = (const float*)d_in[4];
    const float* c2w     = (const float*)d_in[5];
    const float* c2b     = (const float*)d_in[6];
    const float* W_pre   = (const float*)d_in[7];
    const float* W_suc   = (const float*)d_in[8];
    const int* pre_idx   = (const int*)d_in[9];
    const int* suc_idx   = (const int*)d_in[10];
    const int* n_hop_pre = (const int*)d_in[11];
    const int* n_hop_suc = (const int*)d_in[12];
    float* x = (float*)d_out;                      // x / residual lives in d_out

    const size_t ND   = (size_t)N_NODES * DDIM;                     // 12.8M
    const size_t WDD  = (size_t)RROUNDS * DDIM * DDIM;              // 262144
    const size_t WSDD = (size_t)RROUNDS * SHOPS * DDIM * DDIM;      // 1572864

    char* ws = (char*)d_ws;
    float* temp          = (float*)ws;                              // ND f32
    unsigned short* xb   = (unsigned short*)(ws + ND * sizeof(float)); // ND bf16
    unsigned short* wc   = xb  + ND;
    unsigned short* wc2  = wc  + WDD;
    unsigned short* wp   = wc2 + WDD;
    unsigned short* wsu  = wp  + WSDD;

    dim3 blk(256);
    cvt_bf16<<<(unsigned)((WDD  + 255) / 256), blk, 0, stream>>>(W_ctr,  wc,  (int)WDD);
    cvt_bf16<<<(unsigned)((WDD  + 255) / 256), blk, 0, stream>>>(W_ctr2, wc2, (int)WDD);
    cvt_bf16<<<(unsigned)((WSDD + 255) / 256), blk, 0, stream>>>(W_pre,  wp,  (int)WSDD);
    cvt_bf16<<<(unsigned)((WSDD + 255) / 256), blk, 0, stream>>>(W_suc,  wsu, (int)WSDD);
    init_x  <<<(unsigned)((ND   + 255) / 256), blk, 0, stream>>>(lane_features, x, xb, (int)ND);

    const unsigned gemm_blocks = (MTILES + 3) / 4;   // 782
    const unsigned gn_blocks   = N_NODES / 8;        // 6250

    for (int i = 0; i < RROUNDS; ++i) {
        // temp = x @ W_ctr[i].T  (plain store, overwrites temp)
        gemm_bf16<<<gemm_blocks, blk, 0, stream>>>(
            xb, wc + (size_t)i * DDIM * DDIM, temp, nullptr, nullptr);
        // 12 gather->GEMM->scatter-add message passes
        for (int k = 0; k < SHOPS; ++k) {
            const size_t woff = ((size_t)i * SHOPS + k) * DDIM * DDIM;
            gemm_bf16<<<gemm_blocks, blk, 0, stream>>>(
                xb, wp + woff, temp,
                n_hop_pre + (size_t)k * N_NODES, suc_idx + (size_t)k * N_NODES);
            gemm_bf16<<<gemm_blocks, blk, 0, stream>>>(
                xb, wsu + woff, temp,
                n_hop_suc + (size_t)k * N_NODES, pre_idx + (size_t)k * N_NODES);
        }
        // x1 = relu(GN(temp)) -> bf16
        gn_relu_bf16<<<gn_blocks, blk, 0, stream>>>(
            temp, norm_w + (size_t)i * DDIM, norm_b + (size_t)i * DDIM, xb);
        // temp = x1 @ W_ctr2[i].T
        gemm_bf16<<<gemm_blocks, blk, 0, stream>>>(
            xb, wc2 + (size_t)i * DDIM * DDIM, temp, nullptr, nullptr);
        // x = relu(GN(temp) + x); refresh bf16 mirror
        gn_res_relu<<<gn_blocks, blk, 0, stream>>>(
            temp, c2w + (size_t)i * DDIM, c2b + (size_t)i * DDIM, x, xb);
    }
}
